// BinConv2dEval_34359738368298
// MI455X (gfx1250) — 8-bit matrix operand order repaired and checked by reading, compiled, not run on hardware
//
#include <hip/hip_runtime.h>

typedef int v8i __attribute__((ext_vector_type(8)));

#define N_    32
#define CIN_  128
#define H_    64
#define W_    64
#define COUT_ 256
#define HW_   (H_ * W_)
#define KCH   18
#define XQ_BYTES (N_ * H_ * W_ * CIN_)
#define WBLK_BYTES (KCH * 1024)
#define WQ_BYTES (16 * WBLK_BYTES)

__global__ void pack_x_kernel(const float* __restrict__ x,
                              unsigned char* __restrict__ xq) {
    int idx = blockIdx.x * 256 + threadIdx.x;
    int c = idx % CIN_;
    int t = idx / CIN_;
    int w = t % W_;  t /= W_;
    int h = t % H_;
    int n = t / H_;
    float v = x[((n * CIN_ + c) * H_ + h) * W_ + w];
    xq[idx] = (unsigned char)(v > 0.5f ? 1 : 0);
}

__global__ void pack_w_kernel(const float* __restrict__ wsrc,
                              signed char* __restrict__ wq) {
    int idx  = blockIdx.x * 256 + threadIdx.x;
    int b    = idx & 3;
    int v    = (idx >> 2) & 7;
    int lane = (idx >> 5) & 31;
    int kc   = (idx >> 10) % KCH;
    int blk  = idx / WBLK_BYTES;
    int m    = lane & 15;
    int K    = (v >> 2) * 32 + (lane >> 4) * 16 + (v & 3) * 4 + b;
    int tap  = kc >> 1;
    int kh   = tap / 3, kw = tap % 3;
    int cin  = ((kc & 1) << 6) + K;
    int cout = blk * 16 + m;
    float wv = wsrc[((cout * CIN_ + cin) * 3 + kh) * 3 + kw];
    wq[idx] = (signed char)(int)wv;
}

__global__ void __launch_bounds__(256, 1)
binconv_wmma_kernel(const unsigned char* __restrict__ xq,
                    const signed char*  __restrict__ wq,
                    const float* __restrict__ bias,
                    const float* __restrict__ sign,
                    float* __restrict__ out) {
    const int lane  = threadIdx.x & 31;
    const int wave  = threadIdx.x >> 5;
    const int ptile = blockIdx.x * 8 + wave;
    const int cgrp  = blockIdx.y;
    const int pBase = ptile * 16;
    const int n     = pBase / HW_;
    const int hw    = pBase % HW_;
    const int h0    = hw / W_;
    const int w0    = hw % W_;
    const int col   = lane & 15;
    const int khalf = lane >> 4;
    const int k0    = khalf * 16;

    v8i acc0 = {}, acc1 = {}, acc2 = {}, acc3 = {};

    const signed char* wbase = wq + (size_t)(cgrp * 4) * WBLK_BYTES + lane * 32;

    for (int kc = 0; kc < KCH; ++kc) {
        const int tap    = kc >> 1;
        const int kh     = tap / 3, kw = tap % 3;
        const int cinoff = (kc & 1) << 6;

        v8i bm = {};
        const int hh = h0 + kh - 1;
        const int ww = w0 + col + kw - 1;
        if ((unsigned)hh < (unsigned)H_ && (unsigned)ww < (unsigned)W_) {
            const int4* p = (const int4*)(xq +
                (size_t)(((n * H_ + hh) * W_ + ww) * CIN_ + cinoff + k0));
            int4 lo = p[0];
            int4 hi = p[2];
            bm[0] = lo.x; bm[1] = lo.y; bm[2] = lo.z; bm[3] = lo.w;
            bm[4] = hi.x; bm[5] = hi.y; bm[6] = hi.z; bm[7] = hi.w;
        }

        const signed char* wp = wbase + kc * 1024;
#pragma unroll
        for (int mb = 0; mb < 4; ++mb) {
            const int4* ap = (const int4*)(wp + (size_t)mb * WBLK_BYTES);
            int4 a0 = ap[0];
            int4 a1 = ap[1];
            v8i am;
            am[0] = a0.x; am[1] = a0.y; am[2] = a0.z; am[3] = a0.w;
            am[4] = a1.x; am[5] = a1.y; am[6] = a1.z; am[7] = a1.w;
            if (mb == 0)
                acc0 = __builtin_amdgcn_wmma_i32_16x16x64_iu8(true, am, false, bm, acc0, false, false);
            else if (mb == 1)
                acc1 = __builtin_amdgcn_wmma_i32_16x16x64_iu8(true, am, false, bm, acc1, false, false);
            else if (mb == 2)
                acc2 = __builtin_amdgcn_wmma_i32_16x16x64_iu8(true, am, false, bm, acc2, false, false);
            else
                acc3 = __builtin_amdgcn_wmma_i32_16x16x64_iu8(true, am, false, bm, acc3, false, false);
        }
    }

    const int mrow0 = khalf * 8;
    const int hwc   = hw + col;
#pragma unroll
    for (int mb = 0; mb < 4; ++mb) {
        v8i acc = (mb == 0) ? acc0 : (mb == 1) ? acc1 : (mb == 2) ? acc2 : acc3;
#pragma unroll
        for (int r = 0; r < 8; ++r) {
            const int cout = cgrp * 64 + mb * 16 + mrow0 + r;
            const float v  = (float)acc[r] + bias[cout];
            const float s  = sign[cout];
            const float res = (s > 0.0f) ? (v >= 0.0f ? 1.0f : 0.0f)
                                         : (v <= 0.0f ? 1.0f : 0.0f);
            out[(size_t)(n * COUT_ + cout) * HW_ + hwc] = res;
        }
    }
}

extern "C" void kernel_launch(void* const* d_in, const int* in_sizes, int n_in,
                              void* d_out, int out_size, void* d_ws, size_t ws_size,
                              hipStream_t stream) {
    (void)in_sizes; (void)n_in; (void)out_size; (void)ws_size;

    const float* x      = (const float*)d_in[0];
    const float* weight = (const float*)d_in[1];
    const float* bias   = (const float*)d_in[2];
    const float* sign   = (const float*)d_in[3];
    float* out          = (float*)d_out;

    unsigned char* xq = (unsigned char*)d_ws;
    signed char*   wq = (signed char*)d_ws + XQ_BYTES;

    pack_x_kernel<<<XQ_BYTES / 256, 256, 0, stream>>>(x, xq);
    pack_w_kernel<<<WQ_BYTES / 256, 256, 0, stream>>>(weight, wq);

    dim3 grid(8192 / 8, 4);
    binconv_wmma_kernel<<<grid, 256, 0, stream>>>(xq, wq, bias, sign, out);
}
